// MultiHeadAttention_67413806678053
// MI455X (gfx1250) — matrix operand order repaired and checked by reading, not compiled, not run on hardware
//
#include <hip/hip_runtime.h>

typedef __attribute__((ext_vector_type(16))) _Float16 v16h;
typedef __attribute__((ext_vector_type(8)))  _Float16 v8h;
typedef __attribute__((ext_vector_type(8)))  float    v8f;
typedef __attribute__((ext_vector_type(4)))  int      vint4;

constexpr int Bc = 4, Sc = 2048, Hc = 1024, NHc = 16, Dc = 64;
constexpr float INV_SCALE = 0.125f;  // 1/sqrt(D=64)

#define GLOBAL_AS __attribute__((address_space(1)))
#define LDS_AS    __attribute__((address_space(3)))

// ---------------------------------------------------------------- helpers ---

__device__ inline v8f wmma_f16(v16h a, v16h b, v8f c) {
  // D = A(16x32 f16) * B(32x16 f16) + C(16x16 f32)
  return __builtin_amdgcn_wmma_f32_16x16x32_f16(
      /*neg_a=*/false, a, /*neg_b=*/false, b,
      /*c_mod=*/(short)0, c, /*reuse_a=*/false, /*reuse_b=*/false);
}

// Async DMA: 16 bytes per lane, global -> LDS, tracked by ASYNCcnt.
__device__ inline void async_cp16B(const _Float16* g, _Float16* l) {
  __builtin_amdgcn_global_load_async_to_lds_b128(
      (GLOBAL_AS vint4*)(void*)(const_cast<_Float16*>(g)),
      (LDS_AS vint4*)(void*)l, 0, 0);
}

// A-fragment (16x32 f16, per-lane rows) from a row-major [16 x ld] f16 tile.
// Lane L holds row (L&15); ISA k-map: k = (i/8)*16 + (L/16)*8 + (i%8)
// -> two contiguous 16-byte chunks per lane.
__device__ inline v16h ld_frag_a(const _Float16* tile, int lane, int ld, int koff) {
  const _Float16* p = tile + (lane & 15) * ld + koff + ((lane >> 4) << 3);
  v8h lo = *(const v8h*)p;
  v8h hi = *(const v8h*)(p + 16);
  v16h r;
#pragma unroll
  for (int i = 0; i < 8; ++i) { r[i] = lo[i]; r[i + 8] = hi[i]; }
  return r;
}

// B-fragment (32x16 f16, column-major per-lane) from a K-major [N x ld] tile
// (i.e. tile holds B^T rows). Lane L holds col N=(L&15); k = (i/8)*16 + (L/16)*8 + (i%8)
// -> two contiguous 16-byte chunks per lane.
__device__ inline v16h ld_frag_b(const _Float16* tile, int lane, int nrow, int koff, int ld) {
  const _Float16* p = tile + (size_t)(nrow + (lane & 15)) * ld + koff + ((lane >> 4) << 3);
  v8h lo = *(const v8h*)p;
  v8h hi = *(const v8h*)(p + 16);
  v16h r;
#pragma unroll
  for (int i = 0; i < 8; ++i) { r[i] = lo[i]; r[i + 8] = hi[i]; }
  return r;
}

// stage 16 elements into LDS, converting fp32 -> f16
__device__ inline void stage16(const float* __restrict__ src, _Float16* __restrict__ dst) {
  const float4* s4 = (const float4*)src;
#pragma unroll
  for (int i = 0; i < 4; ++i) {
    const float4 x = s4[i];
    dst[i * 4 + 0] = (_Float16)x.x; dst[i * 4 + 1] = (_Float16)x.y;
    dst[i * 4 + 2] = (_Float16)x.z; dst[i * 4 + 3] = (_Float16)x.w;
  }
}
__device__ inline void stage16(const _Float16* __restrict__ src, _Float16* __restrict__ dst) {
  const v8h* s = (const v8h*)src;
  v8h* d = (v8h*)dst;
  d[0] = s[0]; d[1] = s[1];
}

// -------------------------------------------------- weight transpose + cvt ---
// wt[n*K + k] = (f16) w[k*N + n]   (K = N = 1024)
__global__ __launch_bounds__(256) void wcvt_t(const float* __restrict__ w,
                                              _Float16* __restrict__ wt) {
  __shared__ float tile[32][33];
  const int k0 = blockIdx.x * 32, n0 = blockIdx.y * 32;
  const int tx = threadIdx.x & 31, ty = threadIdx.x >> 5;
#pragma unroll
  for (int i = 0; i < 32; i += 8)
    tile[ty + i][tx] = w[(size_t)(k0 + ty + i) * Hc + n0 + tx];
  __syncthreads();
#pragma unroll
  for (int i = 0; i < 32; i += 8)
    wt[(size_t)(n0 + ty + i) * Hc + k0 + tx] = (_Float16)tile[tx][ty + i];
}

// ----------------------------------------------------------------- GEMM -----
// C[64x64 tile] = A[M x K, row-major, fp32 or f16] * Wt[N x K]^T + bias
// MODE 0: f16 out, head layout [B,NH,S,D]   (Q, K projections)
// MODE 1: f16 out, head layout [B,NH,D,S]   (V projection, pre-transposed)
// MODE 2: f32 out, plain [M,N]              (output projection -> d_out)
template <typename AT, int MODE>
__global__ __launch_bounds__(128) void gemm_tn(const AT* __restrict__ A,
                                               const _Float16* __restrict__ Wt,
                                               const float* __restrict__ bias,
                                               void* __restrict__ out) {
  constexpr int K = Hc, N = Hc;
  __shared__ _Float16 As[64 * 32];
  __shared__ _Float16 Bs[64 * 32];
  const int n0 = blockIdx.x * 64;
  const int m0 = blockIdx.y * 64;
  const int t = threadIdx.x, wave = t >> 5, lane = t & 31;
  const int sr = t >> 1, sseg = (t & 1) * 16;

  v8f acc[4] = {};

  for (int k0 = 0; k0 < K; k0 += 32) {
    stage16(A  + (size_t)(m0 + sr) * K + k0 + sseg, &As[sr * 32 + sseg]);
    stage16(Wt + (size_t)(n0 + sr) * K + k0 + sseg, &Bs[sr * 32 + sseg]);
    __syncthreads();
    const v16h a = ld_frag_a(As + wave * 16 * 32, lane, 32, 0);
#pragma unroll
    for (int nb = 0; nb < 4; ++nb) {
      const v16h b = ld_frag_b(Bs, lane, nb * 16, 0, 32);
      acc[nb] = wmma_f16(a, b, acc[nb]);
    }
    __syncthreads();
  }

#pragma unroll
  for (int nb = 0; nb < 4; ++nb) {
    const int n = n0 + nb * 16 + (lane & 15);
    const float bv = bias[n];
#pragma unroll
    for (int r = 0; r < 8; ++r) {
      const int m = m0 + wave * 16 + ((lane >> 4) << 3) + r;
      const float val = acc[nb][r] + bv;
      if (MODE == 0) {
        const int b = m >> 11, s = m & (Sc - 1);
        const int h = n >> 6,  d = n & (Dc - 1);
        ((_Float16*)out)[(((size_t)(b * NHc + h)) * Sc + s) * Dc + d] = (_Float16)val;
      } else if (MODE == 1) {
        const int b = m >> 11, s = m & (Sc - 1);
        const int h = n >> 6,  d = n & (Dc - 1);
        ((_Float16*)out)[(((size_t)(b * NHc + h)) * Dc + d) * Sc + s] = (_Float16)val;
      } else {
        ((float*)out)[(size_t)m * N + n] = val;
      }
    }
  }
}

// ------------------------------------------------------- flash attention ----
// One block = 4 waves = one (b,h) x 64-query tile. Online softmax over key
// tiles of 64. Qh,Kh: [B,NH,S,D] f16 (K-major). Vt: [B,NH,D,S] f16 (K-major
// as B-operand of P*V). O: [B,S,H] f16.
// K/V tiles are double-buffered in LDS and fetched with async-to-LDS DMA
// (ASYNCcnt) so the tile i+1 fetch overlaps tile i's 16 WMMAs.
__global__ __launch_bounds__(128) void flash_attn(const _Float16* __restrict__ Qh,
                                                  const _Float16* __restrict__ Kh,
                                                  const _Float16* __restrict__ Vt,
                                                  _Float16* __restrict__ O) {
  __shared__ _Float16 Qs[64 * 64];
  __shared__ _Float16 Ks[2][64 * 64];
  __shared__ _Float16 Vs[2][64 * 64];
  __shared__ _Float16 Ps[4 * 16 * 64];
  const int bh = blockIdx.y;           // b*NH + h
  const int q0 = blockIdx.x * 64;
  const _Float16* Qp = Qh + (size_t)bh * Sc * Dc;
  const _Float16* Kp = Kh + (size_t)bh * Sc * Dc;
  const _Float16* Vp = Vt + (size_t)bh * Dc * Sc;
  const int t = threadIdx.x, wave = t >> 5, lane = t & 31;

  // issue async DMA: K tile is 64 contiguous [S,D] rows = one 8KB run;
  // V tile is 64 rows of 128B from the [D,S] layout.
  auto issue_k = [&](int kt, int buf) {
#pragma unroll
    for (int i = 0; i < 4; ++i) {
      const int c = t + 128 * i;                         // 16B chunk id, 0..511
      async_cp16B(Kp + (size_t)kt * Dc + c * 8, &Ks[buf][c * 8]);
    }
  };
  auto issue_v = [&](int kt, int buf) {
#pragma unroll
    for (int i = 0; i < 4; ++i) {
      const int c = t + 128 * i;
      const int row = c >> 3, off = (c & 7) * 8;
      async_cp16B(Vp + (size_t)row * Sc + kt + off, &Vs[buf][row * 64 + off]);
    }
  };

  // prologue: Q tile (contiguous 8KB) + first K/V tiles via async DMA
#pragma unroll
  for (int i = 0; i < 4; ++i) {
    const int c = t + 128 * i;
    async_cp16B(Qp + (size_t)q0 * Dc + c * 8, &Qs[c * 8]);
  }
  issue_k(0, 0);
  issue_v(0, 0);
  __builtin_amdgcn_s_wait_asynccnt(0);
  __syncthreads();

  const v16h qf0 = ld_frag_a(Qs + wave * 16 * 64, lane, 64, 0);
  const v16h qf1 = ld_frag_a(Qs + wave * 16 * 64, lane, 64, 32);

  float m_r[8], l_r[8];
  v8f o_acc[4] = {};
#pragma unroll
  for (int r = 0; r < 8; ++r) { m_r[r] = -1e30f; l_r[r] = 0.0f; }

  constexpr int NT = Sc / 64;
  for (int it = 0; it < NT; ++it) {
    const int buf = it & 1;
    // prefetch next tile into the other buffer (freed by last iter's barrier)
    if (it + 1 < NT) {
      issue_k((it + 1) * 64, buf ^ 1);
      issue_v((it + 1) * 64, buf ^ 1);
    }

    // S tile = Q(16x64) * K_tile^T(64x64) per wave: 8 WMMAs
    v8f s_acc[4] = {};
#pragma unroll
    for (int nb = 0; nb < 4; ++nb) {
      s_acc[nb] = wmma_f16(qf0, ld_frag_b(Ks[buf], lane, nb * 16, 0,  64), s_acc[nb]);
      s_acc[nb] = wmma_f16(qf1, ld_frag_b(Ks[buf], lane, nb * 16, 32, 64), s_acc[nb]);
    }

    // online softmax: row M lives in acc element r across one 16-lane half
#pragma unroll
    for (int r = 0; r < 8; ++r) {
      float tm = -1e30f;
#pragma unroll
      for (int nb = 0; nb < 4; ++nb) {
        s_acc[nb][r] *= INV_SCALE;
        tm = fmaxf(tm, s_acc[nb][r]);
      }
#pragma unroll
      for (int msk = 1; msk < 16; msk <<= 1) tm = fmaxf(tm, __shfl_xor(tm, msk, 32));
      const float mn = fmaxf(m_r[r], tm);
      const float corr = __expf(m_r[r] - mn);
      m_r[r] = mn;
      float rs = 0.0f;
#pragma unroll
      for (int nb = 0; nb < 4; ++nb) {
        const float p = __expf(s_acc[nb][r] - mn);
        s_acc[nb][r] = p;
        rs += p;
      }
#pragma unroll
      for (int msk = 1; msk < 16; msk <<= 1) rs += __shfl_xor(rs, msk, 32);
      l_r[r] = l_r[r] * corr + rs;
#pragma unroll
      for (int nb = 0; nb < 4; ++nb) o_acc[nb][r] *= corr;
    }

    // restage P (C-layout) as f16 A-fragments through this wave's LDS slice
    _Float16* pw = Ps + wave * 16 * 64;
    const int prow_hi = (lane >> 4) << 3;
    const int pcol = lane & 15;
#pragma unroll
    for (int nb = 0; nb < 4; ++nb)
#pragma unroll
      for (int r = 0; r < 8; ++r)
        pw[(prow_hi + r) * 64 + nb * 16 + pcol] = (_Float16)s_acc[nb][r];

    // O += P(16x64) * V_tile(64x64): 8 WMMAs
#pragma unroll
    for (int ks = 0; ks < 2; ++ks) {
      const v16h pa = ld_frag_a(pw, lane, 64, ks * 32);
#pragma unroll
      for (int nb = 0; nb < 4; ++nb)
        o_acc[nb] = wmma_f16(pa, ld_frag_b(Vs[buf], lane, nb * 16, ks * 32, 64), o_acc[nb]);
    }

    // next tile's DMA done + all waves done reading this buffer
    __builtin_amdgcn_s_wait_asynccnt(0);
    __syncthreads();
  }

  const int b = bh >> 4, h = bh & 15;
#pragma unroll
  for (int nb = 0; nb < 4; ++nb) {
    const int d = nb * 16 + (lane & 15);
#pragma unroll
    for (int r = 0; r < 8; ++r) {
      const int s = q0 + wave * 16 + ((lane >> 4) << 3) + r;
      const float val = o_acc[nb][r] / l_r[r];
      O[((size_t)(b * Sc + s)) * Hc + h * Dc + d] = (_Float16)val;
    }
  }
}

// ----------------------------------------------------------------- launch ---
extern "C" void kernel_launch(void* const* d_in, const int* in_sizes, int n_in,
                              void* d_out, int out_size, void* d_ws, size_t ws_size,
                              hipStream_t stream) {
  (void)in_sizes; (void)n_in; (void)out_size; (void)ws_size;
  const float* q  = (const float*)d_in[0];
  const float* k  = (const float*)d_in[1];
  const float* v  = (const float*)d_in[2];
  const float* wq = (const float*)d_in[3];
  const float* bq = (const float*)d_in[4];
  const float* wk = (const float*)d_in[5];
  const float* bk = (const float*)d_in[6];
  const float* wv = (const float*)d_in[7];
  const float* bv = (const float*)d_in[8];
  const float* wo = (const float*)d_in[9];
  const float* bo = (const float*)d_in[10];

  char* ws = (char*)d_ws;
  const size_t MB = 1024 * 1024;
  _Float16* wqt = (_Float16*)(ws + 0 * MB);   // 2 MB each
  _Float16* wkt = (_Float16*)(ws + 2 * MB);
  _Float16* wvt = (_Float16*)(ws + 4 * MB);
  _Float16* wot = (_Float16*)(ws + 6 * MB);
  _Float16* Qh  = (_Float16*)(ws + 8 * MB);   // 16 MB each
  _Float16* Kh  = (_Float16*)(ws + 24 * MB);
  _Float16* Vt  = (_Float16*)(ws + 40 * MB);
  _Float16* Ao  = (_Float16*)(ws + 56 * MB);  // attention output [B,S,H] f16

  wcvt_t<<<dim3(32, 32), dim3(256), 0, stream>>>(wq, wqt);
  wcvt_t<<<dim3(32, 32), dim3(256), 0, stream>>>(wk, wkt);
  wcvt_t<<<dim3(32, 32), dim3(256), 0, stream>>>(wv, wvt);
  wcvt_t<<<dim3(32, 32), dim3(256), 0, stream>>>(wo, wot);

  const dim3 gb(128), gg(16, 128);  // x = N/64, y = M/64 (M = B*S = 8192)
  gemm_tn<float, 0><<<gg, gb, 0, stream>>>(q, wqt, bq, Qh);
  gemm_tn<float, 0><<<gg, gb, 0, stream>>>(k, wkt, bk, Kh);
  gemm_tn<float, 1><<<gg, gb, 0, stream>>>(v, wvt, bv, Vt);

  flash_attn<<<dim3(Sc / 64, Bc * NHc), dim3(128), 0, stream>>>(Qh, Kh, Vt, Ao);

  gemm_tn<_Float16, 2><<<gg, gb, 0, stream>>>(Ao, wot, bo, d_out);
}
